// sLSTMCell_70042326663308
// MI455X (gfx1250) — matrix operand order repaired and checked by reading, not compiled, not run on hardware
//
#include <hip/hip_runtime.h>
#include <math.h>

// sLSTM cell, B=8192, DIM=2048.
// gates = [x,h][B,4096] @ W^T[4096, 4*2048] + b  -> fused bf16 WMMA GEMM with
// double-buffered LDS pipeline; epilogue does stabilized sLSTM math in-register.

#define B_SZ 8192
#define DIMC 2048
#define K_SZ (2 * DIMC)   // 4096
#define MT   128          // batch rows per workgroup (8 waves x 16)
#define NT   32           // d-columns per workgroup (per gate)
#define KT   32           // K per WMMA step (bf16 shape 16x16x32)

typedef unsigned int   u32;
typedef unsigned short u16;
typedef __attribute__((ext_vector_type(16))) __bf16 v16bf;
typedef __attribute__((ext_vector_type(4)))  __bf16 v4bf;
typedef __attribute__((ext_vector_type(8)))  float  v8f;
typedef __attribute__((ext_vector_type(4)))  u32    u32x4;
typedef __attribute__((ext_vector_type(4)))  float  f32x4;

union Frag { u32x4 q[2]; v16bf v; };   // 32 bytes: 16 bf16 per lane

// Native converts -> hardware v_cvt_*bf16_f32 path (RNE), not manual bit math.
__device__ __forceinline__ v4bf cvt4(f32x4 v) {
  v4bf r;
  r.x = (__bf16)v.x; r.y = (__bf16)v.y; r.z = (__bf16)v.z; r.w = (__bf16)v.w;
  return r;
}

__global__ __launch_bounds__(256) void slstm_wmma_kernel(
    const float* __restrict__ x,  const float* __restrict__ h,
    const float* __restrict__ c,  const float* __restrict__ n,
    const float* __restrict__ m,
    const float* __restrict__ Wi, const float* __restrict__ bi,
    const float* __restrict__ Wf, const float* __restrict__ bfp,
    const float* __restrict__ Wo, const float* __restrict__ bo,
    const float* __restrict__ Wz, const float* __restrict__ bz,
    float* __restrict__ out)
{
  __shared__ __align__(16) u16 ldsA[2][MT][KT];       // 2 x 8 KB activations
  __shared__ __align__(16) u16 ldsW[2][4][NT][KT];    // 2 x 8 KB weights

  const int tid  = threadIdx.x;
  const int lane = tid & 31;
  const int wave = tid >> 5;          // 0..7 -> 16-row slice of the 128-row tile
  const int d0   = blockIdx.x * NT;   // gate-column tile
  const int m0   = blockIdx.y * MT;   // batch-row tile

  const float* Wg[4] = { Wi, Wf, Wo, Wz };

  v8f zero = {0.f, 0.f, 0.f, 0.f, 0.f, 0.f, 0.f, 0.f};
  v8f acc[4][2];
  #pragma unroll
  for (int g = 0; g < 4; ++g)
    #pragma unroll
    for (int t = 0; t < 2; ++t)
      acc[g][t] = zero;

  // Staging coordinates (per thread)
  const int rowW = tid >> 3;          // 0..31 weight column-in-tile
  const int kqW  = (tid & 7) << 2;    // fp32 quad offset inside 32-wide K row

  f32x4 rA[4], rW[4];                 // in-flight global data for next tile

  auto loadGlobals = [&](int kb) {
    const float* srcA = (kb < DIMC) ? x : h;
    const int kA = (kb < DIMC) ? kb : (kb - DIMC);
    #pragma unroll
    for (int j = 0; j < 4; ++j) {
      int idx = tid + j * 256;
      int row = idx >> 3;
      int kq  = (idx & 7) << 2;
      rA[j] = *(const f32x4*)(srcA + (size_t)(m0 + row) * DIMC + kA + kq);
    }
    #pragma unroll
    for (int g = 0; g < 4; ++g)
      rW[g] = *(const f32x4*)(Wg[g] + (size_t)(d0 + rowW) * K_SZ + kb + kqW);
  };

  auto cvtStore = [&](int p) {
    #pragma unroll
    for (int j = 0; j < 4; ++j) {
      int idx = tid + j * 256;
      int row = idx >> 3;
      int kq  = (idx & 7) << 2;
      *(v4bf*)&ldsA[p][row][kq] = cvt4(rA[j]);
    }
    #pragma unroll
    for (int g = 0; g < 4; ++g)
      *(v4bf*)&ldsW[p][g][rowW][kqW] = cvt4(rW[g]);
  };

  // Fragment gather coordinates (wave32 WMMA VGPR striping):
  // A 16x32 bf16: lanes 0-15 row M=lane hold K{0..7,16..23}; lanes 16-31 K{8..15,24..31}
  // B 32x16 bf16: lanes 0-15 col N=lane hold K{0..7,16..23}; lanes 16-31 K{8..15,24..31}
  const int mrow = wave * 16 + (lane & 15);
  const int kh   = (lane < 16) ? 0 : 8;
  const int ko   = (lane < 16) ? 0 : 16;
  const int nc   = lane & 15;

  // ---- pipeline prologue: stage tile 0 ----
  loadGlobals(0);
  cvtStore(0);
  __syncthreads();

  int p = 0;
  for (int kb = 0; kb < K_SZ; kb += KT) {
    const bool more = (kb + KT) < K_SZ;

    // Next tile's global loads first: consumed only after the WMMA block, so
    // their latency hides behind matrix math.
    if (more) loadGlobals(kb + KT);

    // L2 prefetch two tiles ahead (gfx1250 global_prefetch_b8)
    if (kb + 2 * KT < K_SZ) {
      const int kn = kb + 2 * KT;
      const float* nsrc = (kn < DIMC) ? x : h;
      const int kNN = (kn < DIMC) ? kn : (kn - DIMC);
      __builtin_prefetch(nsrc + (size_t)(m0 + rowW * 4) * DIMC + kNN, 0, 1);
      __builtin_prefetch(Wg[0] + (size_t)(d0 + rowW) * K_SZ + kn + kqW, 0, 1);
    }

    // Gather ALL fragments into distinct live registers, then fence the
    // scheduler so the 8 WMMAs issue back-to-back behind ONE dscnt wait.
    Frag a;
    a.q[0] = *(const u32x4*)&ldsA[p][mrow][kh];
    a.q[1] = *(const u32x4*)&ldsA[p][mrow][kh + 16];
    Frag b[4][2];
    #pragma unroll
    for (int g = 0; g < 4; ++g)
      #pragma unroll
      for (int t = 0; t < 2; ++t) {
        b[g][t].q[0] = *(const u32x4*)&ldsW[p][g][t * 16 + nc][kh];
        b[g][t].q[1] = *(const u32x4*)&ldsW[p][g][t * 16 + nc][kh + 16];
      }
    __builtin_amdgcn_sched_barrier(0);   // keep all ds_loads above the WMMAs

    #pragma unroll
    for (int g = 0; g < 4; ++g)
      #pragma unroll
      for (int t = 0; t < 2; ++t)
        acc[g][t] = __builtin_amdgcn_wmma_f32_16x16x32_bf16(
            false, a.v, false, b[g][t].v, (short)0, acc[g][t], false, false);

    // Convert + store next tile into the other LDS buffer (loadcnt wait lands
    // here, after the WMMAs are already in flight).
    if (more) cvtStore(p ^ 1);
    __syncthreads();
    p ^= 1;
  }

  // ---- fused sLSTM epilogue ----
  // Accum layout: lanes 0-15 col N=lane, VGPR j -> row M=j; lanes 16-31 -> row M=8+j.
  const size_t BD = (size_t)B_SZ * DIMC;
  #pragma unroll
  for (int t = 0; t < 2; ++t) {
    const int col = d0 + t * 16 + nc;
    const float biv = bi[col], bfv = bfp[col], bov = bo[col], bzv = bz[col];
    #pragma unroll
    for (int j = 0; j < 8; ++j) {
      const int row = m0 + wave * 16 + ((lane < 16) ? j : j + 8);
      const size_t off = (size_t)row * DIMC + col;
      const float it = acc[0][t][j] + biv;
      const float ft = acc[1][t][j] + bfv;
      const float ot = acc[2][t][j] + bov;
      const float zt = acc[3][t][j] + bzv;
      const float cv = c[off], nv = n[off], mv = m[off];
      const float e2 = __expf(2.0f * zt);            // tanh via one fast exp
      const float zg = (e2 - 1.0f) / (e2 + 1.0f);
      const float mN = fmaxf(ft + mv, it);
      const float iG = __expf(it - mN);
      const float fG = __expf(ft + mv - mN);
      const float cN = fG * cv + iG * zg;
      const float nN = fG * nv + iG;
      const float hN = (1.0f / (1.0f + __expf(-ot))) * (cN / (nN + 1e-6f));
      out[off]          = hN;
      out[BD + off]     = cN;
      out[2 * BD + off] = nN;
      out[3 * BD + off] = mN;
    }
  }
}

extern "C" void kernel_launch(void* const* d_in, const int* in_sizes, int n_in,
                              void* d_out, int out_size, void* d_ws, size_t ws_size,
                              hipStream_t stream) {
  (void)in_sizes; (void)n_in; (void)out_size; (void)d_ws; (void)ws_size;
  dim3 grid(DIMC / NT, B_SZ / MT, 1);   // 64 x 64 workgroups
  dim3 block(256, 1, 1);                // 8 wave32
  slstm_wmma_kernel<<<grid, block, 0, stream>>>(
      (const float*)d_in[0],  (const float*)d_in[1],
      (const float*)d_in[2],  (const float*)d_in[3],
      (const float*)d_in[4],
      (const float*)d_in[5],  (const float*)d_in[6],
      (const float*)d_in[7],  (const float*)d_in[8],
      (const float*)d_in[9],  (const float*)d_in[10],
      (const float*)d_in[11], (const float*)d_in[12],
      (float*)d_out);
}
